// Rnn_layer_30099130810551
// MI455X (gfx1250) — matrix operand order repaired and checked by reading, not compiled, not run on hardware
//
#include <hip/hip_runtime.h>

typedef __attribute__((ext_vector_type(16))) __bf16 v16bf;
typedef __attribute__((ext_vector_type(8)))  float  v8f;
typedef __attribute__((ext_vector_type(4)))  int    v4i;
typedef v4i __attribute__((address_space(1))) gv4i;   // global int4
typedef v4i __attribute__((address_space(3))) lv4i;   // LDS int4

#define VOCAB  10000
#define EMB    100
#define UNITS  64
#define BATCH  4096
#define TSTEPS 128
#define EPAD   128   // EMB padded to multiple of 32 for K-chunks

#if __has_builtin(__builtin_amdgcn_global_load_async_to_lds_b128) && \
    __has_builtin(__builtin_amdgcn_s_wait_asynccnt)
#define HAVE_ASYNC 1
#else
#define HAVE_ASYNC 0
#endif

// ---------------- fast math (branch-free, uses v_exp_f32 / v_rcp_f32) ----------
__device__ inline float fast_exp2(float x) {
#if __has_builtin(__builtin_amdgcn_exp2f)
  return __builtin_amdgcn_exp2f(x);
#else
  return exp2f(x);
#endif
}
__device__ inline float fast_rcp(float x) {
#if __has_builtin(__builtin_amdgcn_rcpf)
  return __builtin_amdgcn_rcpf(x);
#else
  return 1.0f / x;
#endif
}
__device__ inline float fast_tanh(float x) {
  float cx = fminf(8.0f, fmaxf(-8.0f, x));
  float e2 = fast_exp2(cx * 2.88539008177793f);      // exp(2x)
  return 1.0f - 2.0f * fast_rcp(e2 + 1.0f);
}
__device__ inline float fast_sigmoid(float x) {
  return fast_rcp(1.0f + fast_exp2(x * -1.44269504088896f));
}

// float -> bf16 (round to nearest even), returned as raw ushort
__device__ inline unsigned short f2bf(float f) {
  union { float f; unsigned int u; } c; c.f = f;
  unsigned int u = c.u;
  unsigned int r = u + 0x7FFFu + ((u >> 16) & 1u);
  return (unsigned short)(r >> 16);
}

// A-fragment (16x32 bf16) from row-major LDS tile.
// lane half 0 holds K {kbase+0..7, kbase+16..23}, half 1 the other K octets; m = lane%16.
__device__ inline v16bf load_a_frag(const unsigned short* lds, int stride,
                                    int m, int kbase, int half) {
  union { v16bf v; uint4 q[2]; } u;
  const unsigned short* p = lds + m * stride + kbase + half * 8;
  u.q[0] = *(const uint4*)(p);
  u.q[1] = *(const uint4*)(p + 16);
  return u.v;
}

// B-fragment (32x16 bf16) from row-major [K][N] LDS weights (setup only).
__device__ inline v16bf load_b_frag(const unsigned short* lds, int stride,
                                    int n, int kbase, int half) {
  union { v16bf v; unsigned short s[16]; } u;
  const unsigned short* p = lds + (kbase + half * 8) * stride + n;
#pragma unroll
  for (int j = 0; j < 16; ++j) u.s[j] = p[(j + (j >= 8 ? 8 : 0)) * stride];
  return u.v;
}

// Gather 16 embedding rows for step tt into a bf16 LDS tile (pad region untouched).
__device__ inline void gather_x(unsigned short* dst, const int* tokens,
                                const float* emb, int bTile, int tt, int tid) {
  for (int i = tid; i < 16 * EMB; i += 128) {
    int r = i / EMB, k = i - r * EMB;
    int tok = tokens[(long)(bTile + r) * TSTEPS + tt];
    dst[r * EPAD + k] = f2bf(emb[(long)tok * EMB + k]);
  }
}

// ---------------- Layer 1: fused embed-gather + x@W1 + recurrence ----------------
__global__ __launch_bounds__(128) void rnn_layer1_kernel(
    const int*   __restrict__ tokens,  // [B,T]
    const float* __restrict__ emb,     // [VOCAB,EMB]
    const float* __restrict__ W1,      // [EMB,UNITS]
    const float* __restrict__ U1,      // [UNITS,UNITS]
    const float* __restrict__ b1,      // [UNITS]
    unsigned short* __restrict__ h1out // [B,T,UNITS] bf16
) {
  __shared__ __attribute__((aligned(16))) unsigned short sW1[EPAD * UNITS];   // 16 KB
  __shared__ __attribute__((aligned(16))) unsigned short sU1[UNITS * UNITS];  //  8 KB
  __shared__ __attribute__((aligned(16))) unsigned short sX[2][16 * EPAD];    //  8 KB
  __shared__ __attribute__((aligned(16))) unsigned short sH[2][16 * UNITS];   //  4 KB

  const int tid   = threadIdx.x;
  const int lane  = tid & 31;
  const int wave  = tid >> 5;          // 0..3 -> N columns 16w..16w+15
  const int n     = lane & 15;
  const int half  = lane >> 4;
  const int nGlob = wave * 16 + n;
  const int bTile = blockIdx.x * 16;

  // Stage weights to LDS as bf16 (zero-pad K=100..127 of W1)
  for (int i = tid; i < EPAD * UNITS; i += 128) {
    int k = i / UNITS, c = i - k * UNITS;
    sW1[i] = (k < EMB) ? f2bf(W1[k * UNITS + c]) : (unsigned short)0;
  }
  for (int i = tid; i < UNITS * UNITS; i += 128) sU1[i] = f2bf(U1[i]);
  for (int i = tid; i < 2 * 16 * EPAD; i += 128) sX[0][i] = 0;   // pads stay 0
  for (int i = tid; i < 16 * UNITS;    i += 128) sH[0][i] = 0;   // h0 = 0
  gather_x(sX[0], tokens, emb, bTile, 0, tid);                   // tile for t=0
  __syncthreads();

  // Persistent B-fragments in registers
  v16bf bW[4], bU[2];
#pragma unroll
  for (int c = 0; c < 4; ++c) bW[c] = load_b_frag(sW1, UNITS, n, c * 32, half);
#pragma unroll
  for (int c = 0; c < 2; ++c) bU[c] = load_b_frag(sU1, UNITS, n, c * 32, half);
  const float bias = b1[nGlob];

  for (int t = 0; t < TSTEPS; ++t) {
    const int cur = t & 1, nxt = cur ^ 1;
    if (t + 1 < TSTEPS)                      // prefetch next tile into other buffer
      gather_x(sX[nxt], tokens, emb, bTile, t + 1, tid);

    v8f acc = {bias, bias, bias, bias, bias, bias, bias, bias};
#pragma unroll
    for (int c = 0; c < 4; ++c) {            // xw = x @ W1  (K=128 padded)
      v16bf a = load_a_frag(sX[cur], EPAD, n, c * 32, half);
      acc = __builtin_amdgcn_wmma_f32_16x16x32_bf16(false, a, false, bW[c],
                                                    (short)0, acc, false, false);
    }
#pragma unroll
    for (int c = 0; c < 2; ++c) {            // += h_{t-1} @ U1  (K=64)
      v16bf a = load_a_frag(sH[cur], UNITS, n, c * 32, half);
      acc = __builtin_amdgcn_wmma_f32_16x16x32_bf16(false, a, false, bU[c],
                                                    (short)0, acc, false, false);
    }

#pragma unroll
    for (int j = 0; j < 8; ++j) {            // h_t -> other h buffer + global bf16
      int m = j + 8 * half;
      unsigned short hb = f2bf(fast_tanh(acc[j]));
      sH[nxt][m * UNITS + nGlob] = hb;
      h1out[((long)(bTile + m) * TSTEPS + t) * UNITS + nGlob] = hb;
    }
    __syncthreads();                         // single barrier per step
  }
}

// ---------------- Layer 2: h1@W2 + recurrence + sigmoid head ----------------
__device__ inline void load_tile(unsigned short* dstLds, const unsigned short* h1,
                                 int bTile, int tt, int tid) {
#if HAVE_ASYNC
  // 16 rows x 128 B; 128 threads x one async b128 each (DMA into LDS, ASYNCcnt)
  int r = tid >> 3, c = tid & 7;
  const unsigned short* g = h1 + ((long)(bTile + r) * TSTEPS + tt) * UNITS + c * 8;
  __builtin_amdgcn_global_load_async_to_lds_b128(
      (gv4i*)g, (lv4i*)(dstLds + r * UNITS + c * 8), 0, 0);
#else
  for (int i = tid; i < 16 * (UNITS / 2); i += 128) {
    int r = i >> 5, kk = i & 31;
    ((unsigned int*)dstLds)[r * 32 + kk] =
        ((const unsigned int*)h1)[(((long)(bTile + r) * TSTEPS + tt) * UNITS >> 1) + kk];
  }
#endif
}

__global__ __launch_bounds__(128) void rnn_layer2_kernel(
    const unsigned short* __restrict__ h1, // [B,T,UNITS] bf16
    const float* __restrict__ W2,
    const float* __restrict__ U2,
    const float* __restrict__ b2,
    const float* __restrict__ Wd,          // [UNITS]
    const float* __restrict__ bd,          // [1]
    float* __restrict__ out                // [B]
) {
  __shared__ __attribute__((aligned(16))) unsigned short sW2[UNITS * UNITS];
  __shared__ __attribute__((aligned(16))) unsigned short sU2[UNITS * UNITS];
  __shared__ __attribute__((aligned(16))) unsigned short sX[2][16 * UNITS];
  __shared__ __attribute__((aligned(16))) unsigned short sH[2][16 * UNITS];
  __shared__ float sHT[16 * UNITS];

  const int tid   = threadIdx.x;
  const int lane  = tid & 31;
  const int wave  = tid >> 5;
  const int n     = lane & 15;
  const int half  = lane >> 4;
  const int nGlob = wave * 16 + n;
  const int bTile = blockIdx.x * 16;

  for (int i = tid; i < UNITS * UNITS; i += 128) { sW2[i] = f2bf(W2[i]); sU2[i] = f2bf(U2[i]); }
  for (int i = tid; i < 16 * UNITS;    i += 128) sH[0][i] = 0;
  load_tile(sX[0], h1, bTile, 0, tid);
#if HAVE_ASYNC
  __builtin_amdgcn_s_wait_asynccnt(0);
#endif
  __syncthreads();

  v16bf bW[2], bU[2];
#pragma unroll
  for (int c = 0; c < 2; ++c) {
    bW[c] = load_b_frag(sW2, UNITS, n, c * 32, half);
    bU[c] = load_b_frag(sU2, UNITS, n, c * 32, half);
  }
  const float bias = b2[nGlob];

  for (int t = 0; t < TSTEPS; ++t) {
    const int cur = t & 1, nxt = cur ^ 1;
    if (t + 1 < TSTEPS) load_tile(sX[nxt], h1, bTile, t + 1, tid);

    v8f acc = {bias, bias, bias, bias, bias, bias, bias, bias};
#pragma unroll
    for (int c = 0; c < 2; ++c) {            // xw = h1_t @ W2
      v16bf a = load_a_frag(sX[cur], UNITS, n, c * 32, half);
      acc = __builtin_amdgcn_wmma_f32_16x16x32_bf16(false, a, false, bW[c],
                                                    (short)0, acc, false, false);
    }
#pragma unroll
    for (int c = 0; c < 2; ++c) {            // += h_{t-1} @ U2
      v16bf a = load_a_frag(sH[cur], UNITS, n, c * 32, half);
      acc = __builtin_amdgcn_wmma_f32_16x16x32_bf16(false, a, false, bU[c],
                                                    (short)0, acc, false, false);
    }

#pragma unroll
    for (int j = 0; j < 8; ++j) {
      int m = j + 8 * half;
      float hv = fast_tanh(acc[j]);
      sH[nxt][m * UNITS + nGlob] = f2bf(hv);
      if (t == TSTEPS - 1) sHT[m * UNITS + nGlob] = hv;
    }
#if HAVE_ASYNC
    if (t + 1 < TSTEPS) __builtin_amdgcn_s_wait_asynccnt(0);
#endif
    __syncthreads();                         // single barrier per step
  }

  // Head: out[b] = sigmoid(hT . Wd + bd)
  if (tid < 16) {
    float s = bd[0];
#pragma unroll 8
    for (int h = 0; h < UNITS; ++h) s += sHT[tid * UNITS + h] * Wd[h];
    out[bTile + tid] = fast_sigmoid(s);
  }
}

extern "C" void kernel_launch(void* const* d_in, const int* in_sizes, int n_in,
                              void* d_out, int out_size, void* d_ws, size_t ws_size,
                              hipStream_t stream) {
  (void)in_sizes; (void)n_in; (void)out_size; (void)ws_size;
  const int*   tokens = (const int*)  d_in[0];
  const float* emb    = (const float*)d_in[1];
  const float* W1     = (const float*)d_in[2];
  const float* U1     = (const float*)d_in[3];
  const float* b1     = (const float*)d_in[4];
  const float* W2     = (const float*)d_in[5];
  const float* U2     = (const float*)d_in[6];
  const float* b2     = (const float*)d_in[7];
  const float* Wd     = (const float*)d_in[8];
  const float* bd     = (const float*)d_in[9];

  unsigned short* h1ws = (unsigned short*)d_ws;  // [B,T,UNITS] bf16 = 64 MiB

  rnn_layer1_kernel<<<BATCH / 16, 128, 0, stream>>>(tokens, emb, W1, U1, b1, h1ws);
  rnn_layer2_kernel<<<BATCH / 16, 128, 0, stream>>>(h1ws, W2, U2, b2, Wd, bd,
                                                    (float*)d_out);
}
